// DGCNN_cls_55276229099972
// MI455X (gfx1250) — matrix operand order repaired and checked by reading, not compiled, not run on hardware
//
#include <hip/hip_runtime.h>

typedef __attribute__((ext_vector_type(16))) _Float16 v16h;
typedef __attribute__((ext_vector_type(8)))  float    v8f;
typedef unsigned int u32x4 __attribute__((ext_vector_type(4)));
typedef int          i32x8 __attribute__((ext_vector_type(8)));
typedef int          i32x4 __attribute__((ext_vector_type(4)));

#define NEG_SLOPE 0.2f
#define BN_EPS    1e-5f
#define KNN       20
#define NPTS      1024
#define NBATCH    8

#define AS1 __attribute__((address_space(1)))
#define AS3 __attribute__((address_space(3)))

// gfx1250 async / TDM paths, feature-guarded (host pass + older toolchains fall back)
#if defined(__AMDGCN__) && __has_builtin(__builtin_amdgcn_tensor_load_to_lds) && \
    __has_builtin(__builtin_amdgcn_s_wait_tensorcnt)
#define HAVE_TDM 1
#endif
#if defined(__AMDGCN__) && __has_builtin(__builtin_amdgcn_global_load_async_to_lds_b128) && \
    __has_builtin(__builtin_amdgcn_s_wait_asynccnt)
#define HAVE_ASYNC 1
#endif

// ---------------------------------------------------------------------------
// WMMA fragment loaders (A/B: contiguous k order, same for both, not the ISA table order; C/D per ISA)
// A: 16x32 f16, lane L: M=L&15, g=L>>4; half i -> K=g*16+i
// B: 32x16 f16, lane L: N=L&15, g=L>>4; half i -> K=g*16+i  (contiguous 32B load)
// C/D: v8f, lane L: N=L&15; VGPR r: M=r+(L>=16?8:0)
// ---------------------------------------------------------------------------
template<int LD>
__device__ inline v16h load_a_frag(const _Float16* base, int row, int k0, int lane) {
  const int g = lane >> 4;
  const _Float16* p = base + row * LD + k0;
  v16h a;
#pragma unroll
  for (int i = 0; i < 16; ++i) a[i] = p[g * 16 + i];
  return a;
}

__device__ inline v16h load_b_frag(const _Float16* W, int ld, int o0, int k0, int lane) {
  const int g = lane >> 4, n = lane & 15;
  const _Float16* p = W + (size_t)(o0 + n) * ld + k0 + g * 16;
  v16h b;
#pragma unroll
  for (int i = 0; i < 16; ++i) b[i] = p[i];
  return b;
}

__device__ inline float lrelu(float v) { return v >= 0.f ? v : NEG_SLOPE * v; }

#ifdef HAVE_TDM
// 1-D global->LDS tile copy of `ndw` DWORDs via the Tensor Data Mover, with
// optional LDS padding of 1 DWORD every (2<<ivl) DWORDs (bank-conflict stride).
// D# layout per CDNA5 ISA; 6-arg builtin form.
__device__ inline void tdm_load_1d(unsigned ldsByteAddr, const void* gsrc,
                                   unsigned ndw, unsigned ivl, bool pad) {
  unsigned long long ga = (unsigned long long)(size_t)gsrc;
  u32x4 g0;
  g0[0] = 1u;                                           // count=1, user mode
  g0[1] = ldsByteAddr;                                  // lds_addr (bytes)
  g0[2] = (unsigned)(ga & 0xffffffffu);                 // global_addr[31:0]
  g0[3] = (unsigned)((ga >> 32) & 0x01ffffffu)          // global_addr[56:32]
          | 0x80000000u;                                // type=2 ("image")
  i32x8 g1;
  g1[0] = (int)((2u << 16)                              // data_size = 4B
                | (pad ? (1u << 20) : 0u)               // pad_enable
                | ((ivl & 7u) << 22));                  // pad_interval; amount=1 DW
  g1[1] = (int)((ndw & 0xffffu) << 16);                 // tensor_dim0[15:0]
  g1[2] = (int)(((ndw >> 16) & 0xffffu) | (1u << 16));  // tensor_dim0[31:16], tensor_dim1=1
  g1[3] = (int)((ndw & 0xffffu) << 16);                 // tile_dim0 = ndw (1-D tile)
  g1[4] = 0;                                            // tile_dim1/2 unused
  g1[5] = (int)ndw;                                     // tensor_dim0_stride[31:0]
  g1[6] = 0;
  g1[7] = 0;
  i32x4 z4 = {0, 0, 0, 0};                              // groups 2/3 unused (<=2-D)
  i32x8 z8 = {0, 0, 0, 0, 0, 0, 0, 0};
  __builtin_amdgcn_tensor_load_to_lds(g0, g1, z4, z4, z8, 0);
}
#endif

// ---------------------------------------------------------------------------
// prep: transpose x (B,3,N)->(B,N,3) and convert weights to f16 (w1 padded to 32)
// ---------------------------------------------------------------------------
__global__ void prep_kernel(const float* __restrict__ x,
                            const float* __restrict__ w1, const float* __restrict__ w2,
                            const float* __restrict__ w3, const float* __restrict__ w4,
                            const float* __restrict__ w5,
                            float* __restrict__ xp,
                            _Float16* w1h, _Float16* w2h, _Float16* w3h,
                            _Float16* w4h, _Float16* w5h) {
  int i = blockIdx.x * blockDim.x + threadIdx.x;
  if (i < 24576) {
    int b = i / 3072, rem = i % 3072, n = rem / 3, c = rem % 3;
    xp[i] = x[b * 3072 + c * 1024 + n];
    return;
  }
  i -= 24576;
  if (i < 2048) {                                 // w1: (64,6) -> padded (64,32)
    int o = i >> 5, c = i & 31;
    w1h[i] = (_Float16)(c < 6 ? w1[o * 6 + c] : 0.f);
    return;
  }
  i -= 2048;
  if (i < 8192)  { w2h[i] = (_Float16)w2[i]; return; }  i -= 8192;
  if (i < 16384) { w3h[i] = (_Float16)w3[i]; return; }  i -= 16384;
  if (i < 65536) { w4h[i] = (_Float16)w4[i]; return; }  i -= 65536;
  if (i < 524288){ w5h[i] = (_Float16)w5[i]; return; }
}

// ---------------------------------------------------------------------------
// knn: block = 8 query rows (one per wave). Candidate chunks staged in LDS by
// the TDM (with DMA-side padding) when available; row features via async
// global->LDS b128. Then 20x wave-argmin over the 1024 distances.
// ---------------------------------------------------------------------------
template<int C>
__global__ __launch_bounds__(256) void knn_kernel(const float* __restrict__ xin,
                                                  int* __restrict__ idxBuf) {
  extern __shared__ char smem[];
  float* dist = (float*)smem;               // 8*1024
  float* rowF = dist + 8 * 1024;            // 8*C
  float* chF  = rowF + 8 * C;               // 64*(C+1)
  const int tid = threadIdx.x, lane = tid & 31, w = tid >> 5;
  const int bx = blockIdx.x;
  const int b = bx >> 7, n0 = (bx & 127) * 8;

  const float* rsrc = xin + (size_t)(b * NPTS + n0) * C;   // contiguous 8*C floats
#ifdef HAVE_ASYNC
  for (int e = tid; e < (8 * C) / 4; e += 256) {
    __builtin_amdgcn_global_load_async_to_lds_b128(
        (AS1 i32x4*)(size_t)(rsrc + e * 4),
        (AS3 i32x4*)(size_t)(rowF + e * 4), 0, 0);
  }
  __builtin_amdgcn_s_wait_asynccnt(0);
#else
  for (int e = tid; e < 8 * C; e += 256) rowF[e] = rsrc[e];
#endif

#ifdef HAVE_TDM
  constexpr bool USE_TDM = (C == 64) || (C == 128);
  constexpr unsigned IVL = (C == 64) ? 5u : 6u;   // pad every 64 / 128 DWORDs
#endif
  for (int mc0 = 0; mc0 < NPTS; mc0 += 64) {
    __syncthreads();
#ifdef HAVE_TDM
    if (USE_TDM) {
      if (w == 0) {
        tdm_load_1d((unsigned)(size_t)(void*)chF,
                    xin + (size_t)(b * NPTS + mc0) * C, 64u * C, IVL, true);
        __builtin_amdgcn_s_wait_tensorcnt(0);
      }
    } else
#endif
    {
      for (int e = tid; e < 64 * C; e += 256) {
        int j = e / C, c = e % C;
        chF[j * (C + 1) + c] = xin[((size_t)(b * NPTS + mc0 + j)) * C + c];
      }
    }
    __syncthreads();
    for (int j = lane; j < 64; j += 32) {
      float d = 0.f;
#pragma unroll 4
      for (int c = 0; c < C; ++c) {
        float t = chF[j * (C + 1) + c] - rowF[w * C + c];
        d += t * t;
      }
      dist[w * NPTS + mc0 + j] = d;
    }
  }
  __syncthreads();
  const int n = n0 + w;
  for (int kk = 0; kk < KNN; ++kk) {
    float bd = 3.0e38f; int bi = 0x7fffffff;
    for (int t = 0; t < 32; ++t) {
      int m = lane + (t << 5);
      float d = dist[w * NPTS + m];
      if (d < bd || (d == bd && m < bi)) { bd = d; bi = m; }
    }
    for (int off = 16; off > 0; off >>= 1) {
      float od = __shfl_xor(bd, off);
      int   oi = __shfl_xor(bi, off);
      if (od < bd || (od == bd && oi < bi)) { bd = od; bi = oi; }
    }
    dist[w * NPTS + bi] = 3.0e38f;   // all lanes store same value (in-order DS)
    if (lane == 0) idxBuf[((size_t)(b * NPTS + n)) * KNN + kk] = bi;
  }
}

// ---------------------------------------------------------------------------
// Build edge-feature rows (f16) into LDS; per-row n/neighbor hoisted to LDS
// so the /20 and the idx gather happen once per row, not once per element.
// ---------------------------------------------------------------------------
template<int C, int KDP>
__device__ inline void build_edge_rows(_Float16* eA, int* nbrRow, int* nRow,
                                       const float* __restrict__ xin,
                                       const int* __restrict__ idxBuf,
                                       int b, int m0, int rows, int tid) {
  for (int r = tid; r < rows; r += 256) {
    int m = m0 + r;
    int n = m / KNN, kk = m - n * KNN;
    nRow[r] = n;
    nbrRow[r] = idxBuf[((size_t)(b * NPTS + n)) * KNN + kk];
  }
  __syncthreads();
  const int twoC = 2 * C;
  const float* xb = xin + (size_t)b * NPTS * C;
  for (int e = tid; e < rows * KDP; e += 256) {
    int r = e / KDP, c = e % KDP;          // KDP is a power of two
    float v = 0.f;
    if (c < twoC) {
      int n = nRow[r];
      v = (c < C) ? (xb[(size_t)nbrRow[r] * C + c] - xb[(size_t)n * C + c])
                  : xb[(size_t)n * C + (c - C)];
    }
    eA[e] = (_Float16)v;
  }
}

// ---------------------------------------------------------------------------
// edge_stats: WMMA GEMM of a 64-row edge tile vs all O columns; accumulate
// per-channel sum / sum-of-squares of pre-BN h into global accumulators.
// ---------------------------------------------------------------------------
template<int C, int KDP, int O>
__global__ __launch_bounds__(256) void edge_stats_kernel(const float* __restrict__ xin,
    const int* __restrict__ idxBuf, const _Float16* __restrict__ Wh,
    float* __restrict__ gsum, float* __restrict__ gsumsq) {
  extern __shared__ char smem[];
  _Float16* eA = (_Float16*)smem;                                  // 64*KDP
  float* bsum  = (float*)(smem + 64 * KDP * sizeof(_Float16));     // O
  float* bsq   = bsum + O;                                         // O
  int*   nbrRow = (int*)(bsq + O);                                 // 64
  int*   nRow   = nbrRow + 64;                                     // 64
  const int tid = threadIdx.x, lane = tid & 31, w = tid >> 5;
  const int tilesPerB = (NPTS * KNN) / 64;                         // 320
  const int b = blockIdx.x / tilesPerB, m0 = (blockIdx.x % tilesPerB) * 64;

  for (int o = tid; o < 2 * O; o += 256) bsum[o] = 0.f;
  build_edge_rows<C, KDP>(eA, nbrRow, nRow, xin, idxBuf, b, m0, 64, tid);
  __syncthreads();

  const int nT = 4 * (O / 16);
  for (int t = w; t < nT; t += 8) {
    const int rg = t & 3, o0 = (t >> 2) * 16;
    v8f acc = {};
#pragma unroll
    for (int k0 = 0; k0 < KDP; k0 += 32) {
      v16h a  = load_a_frag<KDP>(eA, rg * 16 + (lane & 15), k0, lane);
      v16h bf = load_b_frag(Wh, KDP, o0, k0, lane);
      acc = __builtin_amdgcn_wmma_f32_16x16x32_f16(false, a, false, bf,
                                                   (short)0, acc, false, false);
    }
    float s1 = 0.f, s2 = 0.f;
#pragma unroll
    for (int r = 0; r < 8; ++r) { float v = acc[r]; s1 += v; s2 += v * v; }
    s1 += __shfl_xor(s1, 16);
    s2 += __shfl_xor(s2, 16);
    if (lane < 16) {
      atomicAdd(&bsum[o0 + lane], s1);
      atomicAdd(&bsq [o0 + lane], s2);
    }
  }
  __syncthreads();
  if (tid < O) { atomicAdd(&gsum[tid], bsum[tid]); atomicAdd(&gsumsq[tid], bsq[tid]); }
}

__global__ void edge_finalize_kernel(const float* __restrict__ gsum,
                                     const float* __restrict__ gsumsq,
                                     const float* __restrict__ gamma,
                                     const float* __restrict__ beta,
                                     float* __restrict__ scale, float* __restrict__ bias,
                                     int O, float invCnt) {
  int o = blockIdx.x * blockDim.x + threadIdx.x;
  if (o >= O) return;
  float mean = gsum[o] * invCnt;
  float var  = gsumsq[o] * invCnt - mean * mean;
  float s = gamma[o] * rsqrtf(var + BN_EPS);
  scale[o] = s;
  bias[o]  = beta[o] - mean * s;
}

// ---------------------------------------------------------------------------
// edge_apply: block = 4 points (80 rows, 5 WMMA row-groups). Recompute h via
// WMMA, apply scale/bias + LeakyReLU, max over the 20 neighbors in LDS.
// ---------------------------------------------------------------------------
template<int C, int KDP, int O>
__global__ __launch_bounds__(256) void edge_apply_kernel(const float* __restrict__ xin,
    const int* __restrict__ idxBuf, const _Float16* __restrict__ Wh,
    const float* __restrict__ scale, const float* __restrict__ bias,
    float* __restrict__ xout) {
  extern __shared__ char smem[];
  _Float16* eA = (_Float16*)smem;                                  // 80*KDP
  float* hcol  = (float*)(smem + 80 * KDP * sizeof(_Float16));     // 80*16
  int*   nbrRow = (int*)(hcol + 80 * 16);                          // 80
  int*   nRow   = nbrRow + 80;                                     // 80
  const int tid = threadIdx.x, lane = tid & 31, w = tid >> 5;
  const int tilesPerB = NPTS / 4;                                  // 256
  const int b = blockIdx.x / tilesPerB, p0 = (blockIdx.x % tilesPerB) * 4;
  const int m0 = p0 * KNN;

  build_edge_rows<C, KDP>(eA, nbrRow, nRow, xin, idxBuf, b, m0, 80, tid);
  __syncthreads();

  for (int ot = 0; ot < O / 16; ++ot) {
    const int o0 = ot * 16;
    if (w < 5) {                                  // wave-uniform branch: EXEC all-1s
      v8f acc = {};
#pragma unroll
      for (int k0 = 0; k0 < KDP; k0 += 32) {
        v16h a  = load_a_frag<KDP>(eA, w * 16 + (lane & 15), k0, lane);
        v16h bf = load_b_frag(Wh, KDP, o0, k0, lane);
        acc = __builtin_amdgcn_wmma_f32_16x16x32_f16(false, a, false, bf,
                                                     (short)0, acc, false, false);
      }
      const int col = lane & 15, hi = lane >> 4;
      const float s = scale[o0 + col], t = bias[o0 + col];
#pragma unroll
      for (int r = 0; r < 8; ++r)
        hcol[(w * 16 + hi * 8 + r) * 16 + col] = lrelu(acc[r] * s + t);
    }
    __syncthreads();
    if (tid < 64) {
      int p = tid >> 4, col = tid & 15;
      float mx = -3.0e38f;
#pragma unroll
      for (int j = 0; j < KNN; ++j) mx = fmaxf(mx, hcol[(p * KNN + j) * 16 + col]);
      xout[((size_t)(b * NPTS + p0 + p)) * O + o0 + col] = mx;
    }
    __syncthreads();
  }
}

// ---------------------------------------------------------------------------
// gemm_h5: (8192 x 512) @ (512 x 1024) via WMMA, 64x64 block tiles; A is the
// on-the-fly concat [x1,x2,x3,x4] converted to f16 in LDS. Raw f32 output.
// ---------------------------------------------------------------------------
__global__ __launch_bounds__(256) void gemm_h5_kernel(const float* __restrict__ x1,
    const float* __restrict__ x2, const float* __restrict__ x3,
    const float* __restrict__ x4, const _Float16* __restrict__ w5h,
    float* __restrict__ h5) {
  __shared__ _Float16 aT[64 * 32];
  const int tid = threadIdx.x, lane = tid & 31, w = tid >> 5;
  const int m0 = blockIdx.x * 64, ob0 = blockIdx.y * 64;
  const int rg = w & 3, ot0 = w >> 2;             // tiles (rg, ot0) and (rg, ot0+2)
  v8f acc0 = {}, acc1 = {};

  for (int k0 = 0; k0 < 512; k0 += 32) {
    __syncthreads();
    for (int e = tid; e < 64 * 32; e += 256) {
      int r = e >> 5, c = e & 31, gc = k0 + c;
      size_t row = (size_t)(m0 + r);
      float v;
      if      (gc < 64)  v = x1[row * 64  + gc];
      else if (gc < 128) v = x2[row * 64  + (gc - 64)];
      else if (gc < 256) v = x3[row * 128 + (gc - 128)];
      else               v = x4[row * 256 + (gc - 256)];
      aT[e] = (_Float16)v;
    }
    __syncthreads();
    v16h a   = load_a_frag<32>(aT, rg * 16 + (lane & 15), 0, lane);
    v16h bf0 = load_b_frag(w5h, 512, ob0 + ot0 * 16,       k0, lane);
    v16h bf1 = load_b_frag(w5h, 512, ob0 + (ot0 + 2) * 16, k0, lane);
    acc0 = __builtin_amdgcn_wmma_f32_16x16x32_f16(false, a, false, bf0,
                                                  (short)0, acc0, false, false);
    acc1 = __builtin_amdgcn_wmma_f32_16x16x32_f16(false, a, false, bf1,
                                                  (short)0, acc1, false, false);
  }
  const int col = lane & 15, hi = lane >> 4;
#pragma unroll
  for (int r = 0; r < 8; ++r) {
    size_t row = (size_t)(m0 + rg * 16 + hi * 8 + r);
    h5[row * 1024 + ob0 + ot0 * 16 + col]       = acc0[r];
    h5[row * 1024 + ob0 + (ot0 + 2) * 16 + col] = acc1[r];
  }
}

__global__ void stats_h5_kernel(const float* __restrict__ h5,
                                const float* __restrict__ g5, const float* __restrict__ b5,
                                float* __restrict__ scale, float* __restrict__ bias) {
  int o = blockIdx.x * blockDim.x + threadIdx.x;
  if (o >= 1024) return;
  float s1 = 0.f, s2 = 0.f;
  for (int r = 0; r < NBATCH * NPTS; ++r) {
    __builtin_prefetch(&h5[(size_t)(r + 16) * 1024 + o], 0, 0);
    float v = h5[(size_t)r * 1024 + o];
    s1 += v; s2 += v * v;
  }
  const float inv = 1.f / (NBATCH * NPTS);
  float mean = s1 * inv, var = s2 * inv - mean * mean;
  float s = g5[o] * rsqrtf(var + BN_EPS);
  scale[o] = s; bias[o] = b5[o] - mean * s;
}

__global__ void pool_kernel(const float* __restrict__ h5, const float* __restrict__ scale,
                            const float* __restrict__ bias, float* __restrict__ pooled) {
  int gid = blockIdx.x * blockDim.x + threadIdx.x;
  if (gid >= NBATCH * 1024) return;
  int b = gid >> 10, o = gid & 1023;
  float s = scale[o], t = bias[o];
  float mx = -3.0e38f, sm = 0.f;
  for (int n = 0; n < NPTS; ++n) {
    __builtin_prefetch(&h5[((size_t)(b * NPTS + n + 16)) * 1024 + o], 0, 0);
    float v = lrelu(h5[((size_t)(b * NPTS + n)) * 1024 + o] * s + t);
    mx = fmaxf(mx, v); sm += v;
  }
  pooled[b * 2048 + o]        = mx;
  pooled[b * 2048 + 1024 + o] = sm * (1.f / NPTS);
}

// ---------------------------------------------------------------------------
// head: single-block FC stack (8 rows) with in-block training-mode BN.
// ---------------------------------------------------------------------------
__global__ __launch_bounds__(256) void head_kernel(const float* __restrict__ pooled,
    const float* __restrict__ lw1, const float* __restrict__ g6, const float* __restrict__ b6,
    const float* __restrict__ lw2, const float* __restrict__ lb2,
    const float* __restrict__ g7, const float* __restrict__ b7,
    const float* __restrict__ lw3, const float* __restrict__ lb3,
    float* __restrict__ out) {
  __shared__ float h6[8 * 512];
  __shared__ float h7[8 * 256];
  const int tid = threadIdx.x;

  for (int t = tid; t < 8 * 512; t += 256) {
    int r = t >> 9, o = t & 511;
    float acc = 0.f;
    const float* pr = pooled + r * 2048;
    const float* wr = lw1 + (size_t)o * 2048;
    for (int c = 0; c < 2048; ++c) acc += pr[c] * wr[c];
    h6[t] = acc;
  }
  __syncthreads();
  for (int o = tid; o < 512; o += 256) {
    float s1 = 0.f, s2 = 0.f;
    for (int r = 0; r < 8; ++r) { float v = h6[r * 512 + o]; s1 += v; s2 += v * v; }
    float mean = s1 * 0.125f, var = s2 * 0.125f - mean * mean;
    float s = g6[o] * rsqrtf(var + BN_EPS), t = b6[o] - mean * s;
    for (int r = 0; r < 8; ++r) h6[r * 512 + o] = lrelu(h6[r * 512 + o] * s + t);
  }
  __syncthreads();
  for (int t = tid; t < 8 * 256; t += 256) {
    int r = t >> 8, o = t & 255;
    float acc = lb2[o];
    for (int c = 0; c < 512; ++c) acc += h6[r * 512 + c] * lw2[(size_t)o * 512 + c];
    h7[t] = acc;
  }
  __syncthreads();
  if (tid < 256) {
    int o = tid;
    float s1 = 0.f, s2 = 0.f;
    for (int r = 0; r < 8; ++r) { float v = h7[r * 256 + o]; s1 += v; s2 += v * v; }
    float mean = s1 * 0.125f, var = s2 * 0.125f - mean * mean;
    float s = g7[o] * rsqrtf(var + BN_EPS), t = b7[o] - mean * s;
    for (int r = 0; r < 8; ++r) h7[r * 256 + o] = lrelu(h7[r * 256 + o] * s + t);
  }
  __syncthreads();
  for (int t = tid; t < 8 * 40; t += 256) {
    int r = t / 40, o = t % 40;
    float acc = lb3[o];
    for (int c = 0; c < 256; ++c) acc += h7[r * 256 + c] * lw3[(size_t)o * 256 + c];
    out[t] = acc;
  }
}

// ---------------------------------------------------------------------------
extern "C" void kernel_launch(void* const* d_in, const int* in_sizes, int n_in,
                              void* d_out, int out_size, void* d_ws, size_t ws_size,
                              hipStream_t stream) {
  const float* x   = (const float*)d_in[0];
  const float* w1  = (const float*)d_in[1];
  const float* g1  = (const float*)d_in[2];
  const float* b1  = (const float*)d_in[3];
  const float* w2  = (const float*)d_in[4];
  const float* g2  = (const float*)d_in[5];
  const float* b2  = (const float*)d_in[6];
  const float* w3  = (const float*)d_in[7];
  const float* g3  = (const float*)d_in[8];
  const float* b3  = (const float*)d_in[9];
  const float* w4  = (const float*)d_in[10];
  const float* g4  = (const float*)d_in[11];
  const float* b4  = (const float*)d_in[12];
  const float* w5  = (const float*)d_in[13];
  const float* g5  = (const float*)d_in[14];
  const float* b5  = (const float*)d_in[15];
  const float* lw1 = (const float*)d_in[16];
  const float* g6  = (const float*)d_in[17];
  const float* b6  = (const float*)d_in[18];
  const float* lw2 = (const float*)d_in[19];
  const float* lb2 = (const float*)d_in[20];
  const float* g7  = (const float*)d_in[21];
  const float* b7  = (const float*)d_in[22];
  const float* lw3 = (const float*)d_in[23];
  const float* lb3 = (const float*)d_in[24];
  float* out = (float*)d_out;

  char* ws = (char*)d_ws;
  size_t off = 0;
  auto alloc = [&](size_t bytes) -> void* {
    void* p = ws + off;
    off = (off + bytes + 255) & ~(size_t)255;
    return p;
  };
  float*    xp     = (float*)alloc(24576u * 4);
  float*    x1     = (float*)alloc(524288u * 4);
  float*    x2     = (float*)alloc(524288u * 4);
  float*    x3     = (float*)alloc(1048576u * 4);
  float*    x4     = (float*)alloc(2097152u * 4);
  float*    h5     = (float*)alloc(8388608u * 4);
  float*    pooled = (float*)alloc(16384u * 4);
  int*      idx    = (int*)alloc(163840u * 4);
  float*    gsum   = (float*)alloc(2048u * 4);      // sum[1024] + sumsq[1024]
  float*    gsumsq = gsum + 1024;
  float*    gscale = (float*)alloc(1024u * 4);
  float*    gbias  = (float*)alloc(1024u * 4);
  _Float16* w1h    = (_Float16*)alloc(2048u * 2);
  _Float16* w2h    = (_Float16*)alloc(8192u * 2);
  _Float16* w3h    = (_Float16*)alloc(16384u * 2);
  _Float16* w4h    = (_Float16*)alloc(65536u * 2);
  _Float16* w5h    = (_Float16*)alloc(524288u * 2);

  const float invCnt = 1.f / (float)(NBATCH * NPTS * KNN);
  const int knnBlocks   = NBATCH * (NPTS / 8);        // 1024
  const int statsBlocks = NBATCH * (NPTS * KNN / 64); // 2560
  const int applyBlocks = NBATCH * (NPTS / 4);        // 2048

  prep_kernel<<<(641024 + 255) / 256, 256, 0, stream>>>(x, w1, w2, w3, w4, w5,
                                                        xp, w1h, w2h, w3h, w4h, w5h);

  // ---- edge conv 1: C=3 (KDP=32), O=64 ----
  {
    size_t shKnn = (8 * 1024 + 8 * 3 + 64 * 4) * sizeof(float);
    knn_kernel<3><<<knnBlocks, 256, shKnn, stream>>>(xp, idx);
    (void)hipMemsetAsync(gsum, 0, 2048 * sizeof(float), stream);
    size_t shS = 64 * 32 * sizeof(_Float16) + 2 * 64 * sizeof(float) + 2 * 64 * sizeof(int);
    edge_stats_kernel<3, 32, 64><<<statsBlocks, 256, shS, stream>>>(xp, idx, w1h, gsum, gsumsq);
    edge_finalize_kernel<<<1, 256, 0, stream>>>(gsum, gsumsq, g1, b1, gscale, gbias, 64, invCnt);
    size_t shA = 80 * 32 * sizeof(_Float16) + 80 * 16 * sizeof(float) + 2 * 80 * sizeof(int);
    edge_apply_kernel<3, 32, 64><<<applyBlocks, 256, shA, stream>>>(xp, idx, w1h, gscale, gbias, x1);
  }
  // ---- edge conv 2: C=64 (KDP=128), O=64 ----
  {
    size_t shKnn = (8 * 1024 + 8 * 64 + 64 * 65) * sizeof(float);
    knn_kernel<64><<<knnBlocks, 256, shKnn, stream>>>(x1, idx);
    (void)hipMemsetAsync(gsum, 0, 2048 * sizeof(float), stream);
    size_t shS = 64 * 128 * sizeof(_Float16) + 2 * 64 * sizeof(float) + 2 * 64 * sizeof(int);
    edge_stats_kernel<64, 128, 64><<<statsBlocks, 256, shS, stream>>>(x1, idx, w2h, gsum, gsumsq);
    edge_finalize_kernel<<<1, 256, 0, stream>>>(gsum, gsumsq, g2, b2, gscale, gbias, 64, invCnt);
    size_t shA = 80 * 128 * sizeof(_Float16) + 80 * 16 * sizeof(float) + 2 * 80 * sizeof(int);
    edge_apply_kernel<64, 128, 64><<<applyBlocks, 256, shA, stream>>>(x1, idx, w2h, gscale, gbias, x2);
  }
  // ---- edge conv 3: C=64 (KDP=128), O=128 ----
  {
    size_t shKnn = (8 * 1024 + 8 * 64 + 64 * 65) * sizeof(float);
    knn_kernel<64><<<knnBlocks, 256, shKnn, stream>>>(x2, idx);
    (void)hipMemsetAsync(gsum, 0, 2048 * sizeof(float), stream);
    size_t shS = 64 * 128 * sizeof(_Float16) + 2 * 128 * sizeof(float) + 2 * 64 * sizeof(int);
    edge_stats_kernel<64, 128, 128><<<statsBlocks, 256, shS, stream>>>(x2, idx, w3h, gsum, gsumsq);
    edge_finalize_kernel<<<1, 256, 0, stream>>>(gsum, gsumsq, g3, b3, gscale, gbias, 128, invCnt);
    size_t shA = 80 * 128 * sizeof(_Float16) + 80 * 16 * sizeof(float) + 2 * 80 * sizeof(int);
    edge_apply_kernel<64, 128, 128><<<applyBlocks, 256, shA, stream>>>(x2, idx, w3h, gscale, gbias, x3);
  }
  // ---- edge conv 4: C=128 (KDP=256), O=256 ----
  {
    size_t shKnn = (8 * 1024 + 8 * 128 + 64 * 129) * sizeof(float);
    knn_kernel<128><<<knnBlocks, 256, shKnn, stream>>>(x3, idx);
    (void)hipMemsetAsync(gsum, 0, 2048 * sizeof(float), stream);
    size_t shS = 64 * 256 * sizeof(_Float16) + 2 * 256 * sizeof(float) + 2 * 64 * sizeof(int);
    edge_stats_kernel<128, 256, 256><<<statsBlocks, 256, shS, stream>>>(x3, idx, w4h, gsum, gsumsq);
    edge_finalize_kernel<<<1, 256, 0, stream>>>(gsum, gsumsq, g4, b4, gscale, gbias, 256, invCnt);
    size_t shA = 80 * 256 * sizeof(_Float16) + 80 * 16 * sizeof(float) + 2 * 80 * sizeof(int);
    edge_apply_kernel<128, 256, 256><<<applyBlocks, 256, shA, stream>>>(x3, idx, w4h, gscale, gbias, x4);
  }

  // ---- pointwise conv to 1024 + BN + pooling ----
  gemm_h5_kernel<<<dim3(128, 16), 256, 0, stream>>>(x1, x2, x3, x4, w5h, h5);
  stats_h5_kernel<<<4, 256, 0, stream>>>(h5, g5, b5, gscale, gbias);
  pool_kernel<<<32, 256, 0, stream>>>(h5, gscale, gbias, pooled);

  // ---- FC head ----
  head_kernel<<<1, 256, 0, stream>>>(pooled, lw1, g6, b6, lw2, lb2, g7, b7, lw3, lb3, out);

  (void)in_sizes; (void)n_in; (void)out_size; (void)ws_size;
}
